// QuantizedLinear_18683107737694
// MI455X (gfx1250) — compile-verified, matrix operand order checked by reading, not run on hardware
//
#include <hip/hip_runtime.h>
#include <hip/hip_bf16.h>

// ---------------------------------------------------------------------------
// QuantizedLinear (superfloat-12 weights, 0.5% fp32 outliers) for MI455X.
//
//   out[T,OUT] = quantize(x)[T,IN] @ (scale * where(mask, w_full, sf_dec(enc))).T + bias
//   T=512, IN=4096, OUT=11008
//
// Memory-roofline design:
//  * Weight streams (enc + full + mask ~ 540MB) read from HBM exactly once:
//    grid tiles N only (BM = all 512 rows), decode fused into the GEMM.
//  * x pre-quantized to f16 once into d_ws (4MB); per-block re-reads hit L2.
//  * v_wmma_f32_16x16x32_f16, f32 accumulate.
//  * A-tile (512x64 f16, 64KB) copied HBM/L2 -> LDS by the Tensor Data Mover
//    (tensor_load_to_lds, TENSORcnt), issued by wave 0, overlapping compute.
//  * B-tile decoded in VALU with register double-buffering; branchless
//    (bitwise) outlier select keeps the weight stream as wide b128 loads.
//  * Double-buffered LDS (144KB), one barrier per 64-wide K-step.
//  * global_prefetch_b8 two tiles ahead on the weight streams.
// ---------------------------------------------------------------------------

#define T_DIM   512
#define IN_DIM  4096
#define OUT_DIM 11008

#define BN    64       // output-feature tile per block
#define BK    64       // K tile (two 16x16x32 WMMA steps)
#define BLOCK 1024     // 32 waves

#define SF_MB    11
#define SF_DENOM 2047.0f
#define SF_MAXV  0.99951171875f   // 1 - 2^-11

typedef _Float16 v16h __attribute__((ext_vector_type(16)));
typedef _Float16 v8h  __attribute__((ext_vector_type(8)));
typedef _Float16 v4h  __attribute__((ext_vector_type(4)));
typedef float    v8f  __attribute__((ext_vector_type(8)));
typedef unsigned int v4u __attribute__((ext_vector_type(4)));
typedef unsigned int v8u __attribute__((ext_vector_type(8)));

// Faithful superfloat decode incl. the reference's sign convention:
// decoded = mag * (2*sign - 1)  ->  sign bit SET => +mag, CLEAR => -mag.
__device__ __forceinline__ float sf_dec(int e) {
    float mag = (float)(e & ((1 << SF_MB) - 1)) * (SF_MAXV / SF_DENOM);
    return ((e >> SF_MB) & 1) ? mag : -mag;
}

// Branchless outlier select: both operands consumed unconditionally so the
// compiler cannot sink the global loads into divergent branches.
__device__ __forceinline__ float sel_w(int m, float outlier, float dec) {
    unsigned s = (unsigned)(-(int)(m != 0));
    unsigned r = (__float_as_uint(outlier) & s) | (__float_as_uint(dec) & ~s);
    return __uint_as_float(r);
}

__device__ __forceinline__ void wait_tensorcnt0() {
#if __has_builtin(__builtin_amdgcn_s_wait_tensorcnt)
    __builtin_amdgcn_s_wait_tensorcnt(0);
#else
    asm volatile("s_wait_tensorcnt 0x0" ::: "memory");
#endif
}

// TDM: 2-D tile load, global -> LDS. 2-SGPR-group form (tensors up to 2D).
// D# layout per CDNA5 ISA ch.8 (group0 128b, group1 256b), data_size = 2B.
__device__ __forceinline__ void tdm_load_2d_f16(
    unsigned lds_off,               // byte offset of destination in LDS
    const void* gaddr,              // global address of tile start
    unsigned tensor_d0,             // tensor dim0 length (elements)
    unsigned tensor_d1,             // tensor dim1 length (elements)
    unsigned tile_d0,               // tile dim0 (elements)
    unsigned tile_d1,               // tile dim1 (elements)
    unsigned long long stride_d0)   // tensor dim0 stride (elements)
{
    unsigned long long ga = (unsigned long long)gaddr;
    v4u g0;
    g0[0] = 1u;                                     // count=1, user descriptor
    g0[1] = lds_off;                                // lds_addr [63:32]
    g0[2] = (unsigned)ga;                           // global_addr[31:0]
    g0[3] = (unsigned)((ga >> 32) & 0x01FFFFFFull)  // global_addr[56:32]
          | (2u << 30);                             // type=2 ("image")
    v8u g1;
    g1[0] = 0x00010000u;                            // wg_mask=0, data_size=1 (2B)
    g1[1] = (tensor_d0 & 0xFFFFu) << 16;            // abar_addr=0 | tensor_dim0[15:0]
    g1[2] = (tensor_d0 >> 16) | ((tensor_d1 & 0xFFFFu) << 16);
    g1[3] = (tensor_d1 >> 16) | (tile_d0 << 16);    // tensor_dim1[31:16] | tile_dim0
    g1[4] = tile_d1 & 0xFFFFu;                      // tile_dim1 | tile_dim2=0
    g1[5] = (unsigned)(stride_d0 & 0xFFFFFFFFull);  // dim0_stride[31:0]
    g1[6] = (unsigned)(stride_d0 >> 32) & 0xFFFFu;  // dim0_stride[47:32] | dim1_stride lo = 0
    g1[7] = 0u;                                     // dim1_stride hi = 0
    asm volatile("tensor_load_to_lds %0, %1" :: "s"(g0), "s"(g1) : "memory");
}

// quantize = sf_decode(sf_encode(x)) (same sign-flip convention), 4 elems/thread
__global__ void __launch_bounds__(256)
sf_quantize_x(const float* __restrict__ x, _Float16* __restrict__ xq, int n4) {
    int i = blockIdx.x * blockDim.x + threadIdx.x;
    if (i >= n4) return;
    float4 v = ((const float4*)x)[i];
    float q[4];
    float vv[4] = {v.x, v.y, v.z, v.w};
    #pragma unroll
    for (int j = 0; j < 4; ++j) {
        float c    = fminf(fmaxf(vv[j], -SF_MAXV), SF_MAXV);
        float mant = floorf(fabsf(c) * SF_DENOM / SF_MAXV);
        float mag  = mant / SF_DENOM * SF_MAXV;
        q[j] = (c < 0.0f) ? mag : -mag;    // (2*sign-1): neg -> +mag, pos -> -mag
    }
    v4h o = { (_Float16)q[0], (_Float16)q[1], (_Float16)q[2], (_Float16)q[3] };
    ((v4h*)xq)[i] = o;
}

__global__ void __launch_bounds__(BLOCK)
sf_gemm_wmma(const _Float16* __restrict__ xq,      // [T, IN] f16 (workspace)
             const int*      __restrict__ enc,     // [OUT, IN]
             const float*    __restrict__ wfull,   // [OUT, IN]
             const int*      __restrict__ mask,    // [OUT, IN]
             const float*    __restrict__ scale,   // [OUT]
             const float*    __restrict__ bias,    // [OUT]
             float*          __restrict__ out)     // [T, OUT]
{
    __shared__ __attribute__((aligned(16))) _Float16 As[2][T_DIM * BK]; // 2x64KB
    __shared__ __attribute__((aligned(16))) _Float16 Bs[2][BN * BK];    // 2x 8KB

    const int tid  = threadIdx.x;
    const int lane = tid & 31;
    const int wave = tid >> 5;              // 0..31
    const int m0   = (wave & 7) * 64;       // 64-row strip per wave
    const int n0   = (wave >> 3) * 16;      // 16-col tile per wave (local)
    const int row0 = blockIdx.x * BN;       // global output-feature base

    const int l15 = lane & 15;
    const int ks  = (lane >> 4) * 8;        // K-half select (gfx1250 16b layout)

    v8f acc[4];
    #pragma unroll
    for (int mt = 0; mt < 4; ++mt) acc[mt] = (v8f)(0.0f);

    // ---- B staging coordinates: thread -> 4 consecutive weights of one row
    const int e4i   = tid * 4;
    const int b_row = e4i >> 6;             // 0..63 local weight row
    const int b_k   = e4i & 63;             // multiple of 4
    const float sc  = scale[row0 + b_row];
    const size_t wbase = (size_t)(row0 + b_row) * IN_DIM + b_k;

    // LDS byte offsets of the two A buffers (flat LDS addr keeps offset in [31:0])
    const unsigned asbase[2] = {
        (unsigned)(uintptr_t)(void*)&As[0][0],
        (unsigned)(uintptr_t)(void*)&As[1][0]
    };

    // ---- register prefetch state for B ----
    int4   e_pf;
    float4 f_pf;
    int4   m_pf;

    // prologue: TDM tile 0 -> As[0]; B tile 0 -> regs -> Bs[0]
    if (wave == 0) {
        tdm_load_2d_f16(asbase[0], xq, IN_DIM, T_DIM, BK, T_DIM, IN_DIM);
    }
    e_pf = *(const int4*)  (enc   + wbase);
    f_pf = *(const float4*)(wfull + wbase);
    m_pf = *(const int4*)  (mask  + wbase);
    {
        float w0 = sel_w(m_pf.x, f_pf.x, sf_dec(e_pf.x)) * sc;
        float w1 = sel_w(m_pf.y, f_pf.y, sf_dec(e_pf.y)) * sc;
        float w2 = sel_w(m_pf.z, f_pf.z, sf_dec(e_pf.z)) * sc;
        float w3 = sel_w(m_pf.w, f_pf.w, sf_dec(e_pf.w)) * sc;
        v4h bv = { (_Float16)w0, (_Float16)w1, (_Float16)w2, (_Float16)w3 };
        *(v4h*)&Bs[0][b_row * BK + b_k] = bv;
    }
    if (wave == 0) wait_tensorcnt0();
    __syncthreads();

    int buf = 0;
    for (int kk = 0; kk < IN_DIM; kk += BK) {
        const bool has_next = (kk + BK) < IN_DIM;

        // ---- kick off next tile early: TDM for A, register loads for B ----
        if (has_next) {
            const int kn = kk + BK;
            if (wave == 0) {
                tdm_load_2d_f16(asbase[buf ^ 1], xq + kn,
                                IN_DIM, T_DIM, BK, T_DIM, IN_DIM);
            }
            e_pf = *(const int4*)  (enc   + wbase + kn);
            f_pf = *(const float4*)(wfull + wbase + kn);
            m_pf = *(const int4*)  (mask  + wbase + kn);
        }
        // L2 prefetch of weight streams two tiles ahead (global_prefetch_b8)
        if (kk + 2 * BK < IN_DIM) {
            const size_t pg = wbase + kk + 2 * BK;
            __builtin_prefetch(enc   + pg, 0, 1);
            __builtin_prefetch(wfull + pg, 0, 1);
            __builtin_prefetch(mask  + pg, 0, 1);
        }

        // ---- compute on current buffer: 2 K-steps x 4 M-tiles = 8 WMMA ----
        const _Float16* Ab = As[buf];
        const _Float16* Bb = Bs[buf];
        #pragma unroll
        for (int kb = 0; kb < BK; kb += 32) {
            const int bbase = (n0 + l15) * BK + kb + ks;
            v16h bf;
            ((v8h*)&bf)[0] = *(const v8h*)&Bb[bbase];
            ((v8h*)&bf)[1] = *(const v8h*)&Bb[bbase + 16];
            #pragma unroll
            for (int mt = 0; mt < 4; ++mt) {
                const int abase = (m0 + mt * 16 + l15) * BK + kb + ks;
                v16h af;
                ((v8h*)&af)[0] = *(const v8h*)&Ab[abase];
                ((v8h*)&af)[1] = *(const v8h*)&Ab[abase + 16];
                acc[mt] = __builtin_amdgcn_wmma_f32_16x16x32_f16(
                    false, af, false, bf, (short)0, acc[mt], false, false);
            }
        }

        // ---- stage prefetched B tile into the other buffer ----
        if (has_next) {
            float w0 = sel_w(m_pf.x, f_pf.x, sf_dec(e_pf.x)) * sc;
            float w1 = sel_w(m_pf.y, f_pf.y, sf_dec(e_pf.y)) * sc;
            float w2 = sel_w(m_pf.z, f_pf.z, sf_dec(e_pf.z)) * sc;
            float w3 = sel_w(m_pf.w, f_pf.w, sf_dec(e_pf.w)) * sc;
            v4h bv = { (_Float16)w0, (_Float16)w1, (_Float16)w2, (_Float16)w3 };
            *(v4h*)&Bs[buf ^ 1][b_row * BK + b_k] = bv;
            // A tile for buf^1 arrives via TDM: wave 0 fences it before barrier
            if (wave == 0) wait_tensorcnt0();
        }
        __syncthreads();
        buf ^= 1;
    }

    // ---- epilogue: D layout (lane<16: M=r, N=lane; lane>=16: M=8+r) ----
    const int col  = row0 + n0 + l15;        // global output feature
    const float bv = bias[col];
    #pragma unroll
    for (int mt = 0; mt < 4; ++mt) {
        const int rbase = m0 + mt * 16 + (lane >> 4) * 8;
        #pragma unroll
        for (int r = 0; r < 8; ++r) {
            out[(size_t)(rbase + r) * OUT_DIM + col] = acc[mt][r] + bv;
        }
    }
}

extern "C" void kernel_launch(void* const* d_in, const int* in_sizes, int n_in,
                              void* d_out, int out_size, void* d_ws, size_t ws_size,
                              hipStream_t stream) {
    const float* x      = (const float*)d_in[0];   // [512, 4096]
    const float* wfull  = (const float*)d_in[1];   // [11008, 4096]
    const float* scale  = (const float*)d_in[2];   // [11008]
    const float* bias   = (const float*)d_in[3];   // [11008]
    const int*   enc    = (const int*)d_in[4];     // [11008, 4096]
    const int*   mask   = (const int*)d_in[5];     // [11008, 4096] (bool as int)
    float*       out    = (float*)d_out;           // [512, 11008]

    _Float16* xq = (_Float16*)d_ws;                // 512*4096*2 = 4 MB scratch

    const int n4 = (T_DIM * IN_DIM) / 4;
    sf_quantize_x<<<(n4 + 255) / 256, 256, 0, stream>>>(x, xq, n4);

    sf_gemm_wmma<<<OUT_DIM / BN, BLOCK, 0, stream>>>(
        xq, enc, wfull, mask, scale, bias, out);
}
